// GCNLayer_72988674228320
// MI455X (gfx1250) — matrix operand order repaired and checked by reading, not compiled, not run on hardware
//
#include <hip/hip_runtime.h>

// ---- CDNA5 WMMA vector types --------------------------------------------
typedef _Float16 v16h __attribute__((ext_vector_type(16)));
typedef float    v8f  __attribute__((ext_vector_type(8)));

#define D       128          // feature dim (in == out == 128)
#define LDS_PITCH 144        // halves per W row: 128 + 16 pad (rows stay 16B aligned, banks spread)

// -------------------------------------------------------------------------
// Zero the output (harness poisons d_out with 0xAA; atomics need zeros).
// -------------------------------------------------------------------------
__global__ void gcn_zero_f4(float4* __restrict__ p, int n4) {
    int i = blockIdx.x * blockDim.x + threadIdx.x;
    if (i < n4) p[i] = make_float4(0.f, 0.f, 0.f, 0.f);
}

// -------------------------------------------------------------------------
// H = X * W^T via v_wmma_f32_16x16x32_f16.
// Block = 256 threads = 8 waves. W (128x128 f32) converted to f16 in LDS.
// Each wave computes a 16-row strip of H (8 tiles of 16x16, K = 4 chunks of 32).
// Lane packing used here (contiguous K order, same for A and B):
//   A 16x32 f16 : lane m=L%16, kh=L/16 holds K = 32c + 16kh + [0..15] (contig)
//   B 32x16 f16 : lane n=L%16, kg=L/16 holds K = 32c + 16kg + [0..15] (contig)
//   D 16x16 f32 : VGPR r, lane L -> M = r + 8*(L/16), N = L%16
// -------------------------------------------------------------------------
__global__ void __launch_bounds__(256)
gcn_gemm_wmma(const float* __restrict__ x, const float* __restrict__ w,
              float* __restrict__ h, int nrows) {
    __shared__ _Float16 wlds[D * LDS_PITCH];

    const int tid = threadIdx.x;
    // Cooperative fp32 -> fp16 conversion of W into LDS (coalesced reads).
    for (int i = tid; i < D * D; i += 256) {
        int o = i >> 7, k = i & 127;
        wlds[o * LDS_PITCH + k] = (_Float16)w[i];
    }
    __syncthreads();

    const int wave    = tid >> 5;
    const int lane    = tid & 31;
    const int strip   = blockIdx.x * 8 + wave;   // 16-row strip index
    const int rowbase = strip * 16;
    if (rowbase >= nrows) return;                // wave-uniform: EXEC stays all-1s

    const int m  = lane & 15;   // A row / B col / D col within tile
    const int kh = lane >> 4;   // K half-group selector

    v8f acc[8] = {};            // 8 output tiles of 16x16 f32 (64 VGPRs)

    const float* px = x + (size_t)(rowbase + m) * D;

#pragma unroll
    for (int c = 0; c < 4; ++c) {
        const int kbase = c * 32;

        // ---- A tile: one 16-float run (64B), convert to f16 ----
        const float* pa = px + kbase + kh * 16;
        float4 f0 = *(const float4*)(pa + 0);
        float4 f1 = *(const float4*)(pa + 4);
        float4 f2 = *(const float4*)(pa + 8);
        float4 f3 = *(const float4*)(pa + 12);
        v16h a;
        a[0]  = (_Float16)f0.x; a[1]  = (_Float16)f0.y;
        a[2]  = (_Float16)f0.z; a[3]  = (_Float16)f0.w;
        a[4]  = (_Float16)f1.x; a[5]  = (_Float16)f1.y;
        a[6]  = (_Float16)f1.z; a[7]  = (_Float16)f1.w;
        a[8]  = (_Float16)f2.x; a[9]  = (_Float16)f2.y;
        a[10] = (_Float16)f2.z; a[11] = (_Float16)f2.w;
        a[12] = (_Float16)f3.x; a[13] = (_Float16)f3.y;
        a[14] = (_Float16)f3.z; a[15] = (_Float16)f3.w;

        // ---- 8 B tiles from LDS, 32B contiguous per lane ----
#pragma unroll
        for (int g = 0; g < 8; ++g) {
            const _Float16* pb =
                &wlds[(size_t)(g * 16 + m) * LDS_PITCH + kbase + kh * 16];
            v16h b = *(const v16h*)pb;   // two ds_load_b128
            acc[g] = __builtin_amdgcn_wmma_f32_16x16x32_f16(
                false, a, false, b, (short)0, acc[g], false, false);
        }
    }

    // ---- Store D tiles: lane covers column m, rows rowbase+8*kh+[0..7] ----
    float* hr = h + (size_t)(rowbase + kh * 8) * D + m;
#pragma unroll
    for (int g = 0; g < 8; ++g) {
#pragma unroll
        for (int r = 0; r < 8; ++r) {
            hr[(size_t)r * D + g * 16] = acc[g][r];
        }
    }
}

// -------------------------------------------------------------------------
// Edge scatter: out[dst] += val * H[src]. One wave per edge; lane L covers
// columns {L, L+32, L+64, L+96} so each load/atomic instruction spans a
// contiguous 128B region. H and out live in the 192MB L2; atomics resolve
// at L2 via global_atomic_add_f32.
// -------------------------------------------------------------------------
__global__ void __launch_bounds__(256)
gcn_scatter(const float* __restrict__ h, const long long* __restrict__ ei,
            const float* __restrict__ ev, float* __restrict__ out, int E) {
    const int e = blockIdx.x * 8 + (threadIdx.x >> 5);
    if (e >= E) return;
    const int lane = threadIdx.x & 31;

    const long long dst = ei[e];                  // edge_index row 0
    const long long src = ei[(size_t)E + e];      // edge_index row 1
    const float     v   = ev[e];

    const float* hp = h   + (size_t)src * D;
    float*       op = out + (size_t)dst * D;

#pragma unroll
    for (int j = 0; j < 4; ++j) {
        const int c = j * 32 + lane;
        atomicAdd(op + c, v * hp[c]);
    }
}

// -------------------------------------------------------------------------
extern "C" void kernel_launch(void* const* d_in, const int* in_sizes, int n_in,
                              void* d_out, int out_size, void* d_ws, size_t ws_size,
                              hipStream_t stream) {
    const float*     x  = (const float*)d_in[0];      // [N,128] f32
    const float*     w  = (const float*)d_in[1];      // [128,128] f32
    const long long* ei = (const long long*)d_in[2];  // [2,E] int64
    const float*     ev = (const float*)d_in[3];      // [E] f32

    const int N = in_sizes[0] / D;
    const int E = in_sizes[3];

    float* out = (float*)d_out;
    float* hws = (float*)d_ws;                        // H scratch: N*128 f32

    // 1) zero the accumulation target
    const int n4 = out_size / 4;
    gcn_zero_f4<<<(n4 + 255) / 256, 256, 0, stream>>>((float4*)out, n4);

    // 2) H = X * W^T  (WMMA f16 -> f32)
    gcn_gemm_wmma<<<(N + 127) / 128, 256, 0, stream>>>(x, w, hws, N);

    // 3) out[dst] += a_ij * H[src]  (one wave per edge)
    gcn_scatter<<<(E + 7) / 8, 256, 0, stream>>>(hws, ei, ev, out, E);
}
